// BarMamba_42812234006550
// MI455X (gfx1250) — matrix operand order repaired and checked by reading, not compiled, not run on hardware
//
#include <hip/hip_runtime.h>
#include <hip/hip_bf16.h>
#include <math.h>

// ---------------------------------------------------------------------------
// BarMamba forward for MI455X (gfx1250), wave32 + v_wmma_f32_16x16x32_bf16.
// ---------------------------------------------------------------------------

typedef __bf16 bf16_t;
typedef __attribute__((ext_vector_type(16))) __bf16 v16bf;
typedef __attribute__((ext_vector_type(8)))  __bf16 v8bf;
typedef __attribute__((ext_vector_type(8)))  float  v8f;

#define BATCH    8
#define SEQ      4096
#define DIM      512
#define HEADS    8
#define PBLK     32
#define NBLK     128          // SEQ / PBLK
#define DH       64           // DIM / HEADS
#define WL       1024         // WIDTHS[LVL]
#define MEMTOT   8064         // sum(WIDTHS)
#define LVLSTART 6144         // level_start_index[LVL]

// sinusoidal PE: c<256 -> sin(pos*invfreq[c]), else cos(pos*invfreq[c-256])
__device__ __forceinline__ float pe_val(float pos, int c) {
  const float k = 9.210340371976184f / 256.0f;  // ln(10000)/half
  if (c < 256) return __sinf(pos * __expf(-(float)c * k));
  return __cosf(pos * __expf(-(float)(c - 256) * k));
}

// ---- WMMA fragment loaders (contiguous k order k = 16*(L/16) + s for A and B, wave32) ---------
// A (16x32 bf16, row major source): lane L holds row (L&15); K elements
// [kb..kb+15] with kb = (L<16)?0:16.
__device__ __forceinline__ v16bf load_A_frag(const bf16_t* base, int ld,
                                             int m0, int k0, int lane) {
  const int row = m0 + (lane & 15);
  const int kb  = (lane < 16) ? 0 : 16;
  const bf16_t* p = base + (size_t)row * ld + k0 + kb;
  v8bf lo = *(const v8bf*)(p);
  v8bf hi = *(const v8bf*)(p + 8);
  return __builtin_shufflevector(lo, hi, 0,1,2,3,4,5,6,7,8,9,10,11,12,13,14,15);
}

// B (32x16 bf16): lane L holds column (L&15); K = k0 + ((L<16)?0:16) .. +15,
// read from an [n][k] (transposed-weight) row-major array -> 32B contiguous.
__device__ __forceinline__ v16bf load_B_frag(const bf16_t* base, int ld,
                                             int n0, int k0, int lane) {
  const int col = n0 + (lane & 15);
  const int ko  = k0 + ((lane < 16) ? 0 : 16);
  return *(const v16bf*)(base + (size_t)col * ld + ko);
}

// ---------------------------------------------------------------------------
// prep kernels
// ---------------------------------------------------------------------------

// transpose 6 weights (512x512 f32, [k][n]) -> bf16 [w][n][k]
__global__ void prep_w(const float* __restrict__ W0, const float* __restrict__ W1,
                       const float* __restrict__ W2, const float* __restrict__ W3,
                       const float* __restrict__ W4, const float* __restrict__ W5,
                       bf16_t* __restrict__ wt) {
  size_t i = (size_t)blockIdx.x * 256 + threadIdx.x;   // 6*512*512
  int w = (int)(i >> 18);
  int rem = (int)(i & 262143);
  int n = rem >> 9, k = rem & 511;
  const float* W = (w == 0) ? W0 : (w == 1) ? W1 : (w == 2) ? W2
                 : (w == 3) ? W3 : (w == 4) ? W4 : W5;
  wt[i] = (bf16_t)W[(size_t)k * DIM + n];
}

// y + bar PE -> bf16  (pe depends only on block index and channel)
__global__ void prep_ype(const float* __restrict__ y, bf16_t* __restrict__ ype) {
  size_t i = (size_t)blockIdx.x * 256 + threadIdx.x;   // B*S*D
  int c = (int)(i & 511);
  size_t row = i >> 9;
  int s = (int)(row & (SEQ - 1));
  int n = s >> 5;
  float v = y[i] + pe_val((float)(n + 1), c);
  ype[i] = (bf16_t)v;
}

// gather row 0 of each block -> A_q (B*N rows)
__global__ void prep_aq(const bf16_t* __restrict__ ype, bf16_t* __restrict__ aq) {
  size_t i = (size_t)blockIdx.x * 256 + threadIdx.x;   // (B*N)*D
  size_t blk = i >> 9;
  int c = (int)(i & 511);
  aq[i] = ype[(blk * PBLK) * (size_t)DIM + c];
}

// memory[:, 6144:7168, :] -> bf16 (B*WL rows)
__global__ void prep_mem(const float* __restrict__ mem, bf16_t* __restrict__ mb) {
  size_t i = (size_t)blockIdx.x * 256 + threadIdx.x;   // (B*WL)*D
  size_t r = i >> 9;
  int c = (int)(i & 511);
  int b = (int)(r >> 10), w = (int)(r & (WL - 1));
  mb[i] = (bf16_t)mem[((size_t)b * MEMTOT + LVLSTART + w) * DIM + c];
}

__global__ void zero_f32(float* __restrict__ p, int n) {
  int i = blockIdx.x * 256 + threadIdx.x;
  if (i < n) p[i] = 0.0f;
}

// ---------------------------------------------------------------------------
// generic bf16 GEMM: C[r][n] = sum_k A[r][k] * Wt[n][k], K=N=512.
// grid = (rows/32, 4), block = 128 (4 waves). Each wave owns a 32x32 output
// tile = 2M x 2N accumulators -> 4 WMMAs per k-step from 2 A-frags (LDS) and
// 2 B-frags (global, L2-resident transposed weights). Halves operand traffic
// per WMMA vs a 1-N-tile wave and gives the scheduler independent WMMA pairs.
// pe_mode==1 adds time PE (K_mem path).
// ---------------------------------------------------------------------------
__global__ __launch_bounds__(128) void gemm_nt(const bf16_t* __restrict__ A,
                                               const bf16_t* __restrict__ Wt,
                                               bf16_t* __restrict__ Cb,
                                               float* __restrict__ Cf,
                                               int pe_mode) {
  __shared__ bf16_t sA[32 * 520];                 // +8 bf16 pad, keeps 16B align
  const int t = threadIdx.x;
  const size_t rBase = (size_t)blockIdx.x * 32;

  for (int c = t; c < 2048; c += 128) {           // 32 rows x 64 chunks of 8
    int row = c >> 6, off = (c & 63) << 3;
    *(v8bf*)&sA[row * 520 + off] = *(const v8bf*)(A + (rBase + row) * DIM + off);
  }
  __syncthreads();

  const int wave = t >> 5, lane = t & 31;
  const int n0 = blockIdx.y * 128 + wave * 32;    // 32 output columns per wave
  v8f c00 = {0,0,0,0,0,0,0,0};
  v8f c01 = {0,0,0,0,0,0,0,0};
  v8f c10 = {0,0,0,0,0,0,0,0};
  v8f c11 = {0,0,0,0,0,0,0,0};
  for (int kk = 0; kk < 16; ++kk) {
    const int k0 = kk * 32;
    v16bf b0 = load_B_frag(Wt, DIM, n0,      k0, lane);
    v16bf b1 = load_B_frag(Wt, DIM, n0 + 16, k0, lane);
    v16bf a0 = load_A_frag(sA, 520, 0,  k0, lane);
    v16bf a1 = load_A_frag(sA, 520, 16, k0, lane);
    c00 = __builtin_amdgcn_wmma_f32_16x16x32_bf16(false, a0, false, b0,
                                                  (short)0, c00, false, false);
    c01 = __builtin_amdgcn_wmma_f32_16x16x32_bf16(false, a0, false, b1,
                                                  (short)0, c01, false, false);
    c10 = __builtin_amdgcn_wmma_f32_16x16x32_bf16(false, a1, false, b0,
                                                  (short)0, c10, false, false);
    c11 = __builtin_amdgcn_wmma_f32_16x16x32_bf16(false, a1, false, b1,
                                                  (short)0, c11, false, false);
  }

  const int col0 = n0 + (lane & 15);
  const int col1 = col0 + 16;
  const int rlo  = (lane < 16) ? 0 : 8;
  const float tsc = 1024.0f / 1023.0f;            // t_norm * W_l
#pragma unroll
  for (int r = 0; r < 8; ++r) {
    size_t g0 = rBase + rlo + r;
    size_t g1 = g0 + 16;
    float v00 = c00[r], v01 = c01[r], v10 = c10[r], v11 = c11[r];
    if (pe_mode == 1) {
      float p0 = (float)(g0 & (WL - 1)) * tsc;
      float p1 = (float)(g1 & (WL - 1)) * tsc;
      v00 += pe_val(p0, col0); v01 += pe_val(p0, col1);
      v10 += pe_val(p1, col0); v11 += pe_val(p1, col1);
    }
    if (Cb) {
      Cb[g0 * DIM + col0] = (bf16_t)v00; Cb[g0 * DIM + col1] = (bf16_t)v01;
      Cb[g1 * DIM + col0] = (bf16_t)v10; Cb[g1 * DIM + col1] = (bf16_t)v11;
    }
    if (Cf) {
      Cf[g0 * DIM + col0] = v00; Cf[g0 * DIM + col1] = v01;
      Cf[g1 * DIM + col0] = v10; Cf[g1 * DIM + col1] = v11;
    }
  }
}

// ---------------------------------------------------------------------------
// bar attention: 1 workgroup per (b,n) block, 1 wave per head.
// 31 keys -> lane j computes score j; shuffle softmax; ctx -> bf16.
// ---------------------------------------------------------------------------
__global__ __launch_bounds__(256) void attn_bar(const bf16_t* __restrict__ qs,
                                                const bf16_t* __restrict__ kb,
                                                const bf16_t* __restrict__ vb,
                                                bf16_t* __restrict__ ctx) {
  __shared__ float sq[DIM];
  const size_t blk = blockIdx.x;                  // b*NBLK + n
  const int t = threadIdx.x;
  sq[t]       = (float)qs[blk * DIM + t];
  sq[t + 256] = (float)qs[blk * DIM + t + 256];
  __syncthreads();

  const int h = t >> 5, lane = t & 31;
  float s = -INFINITY;
  if (lane < 31) {
    const bf16_t* krow = kb + (blk * PBLK + 1 + lane) * (size_t)DIM + h * DH;
    float acc = 0.0f;
#pragma unroll
    for (int d = 0; d < DH; ++d) acc += sq[h * DH + d] * (float)krow[d];
    s = acc * 0.125f;                             // 1/sqrt(64)
  }
  float m = s;
  for (int o = 16; o > 0; o >>= 1) m = fmaxf(m, __shfl_xor(m, o, 32));
  float p = (lane < 31) ? __expf(s - m) : 0.0f;
  float l = p;
  for (int o = 16; o > 0; o >>= 1) l += __shfl_xor(l, o, 32);
  p /= l;

  float a0 = 0.0f, a1 = 0.0f;                     // dims lane, lane+32
  for (int j = 0; j < 31; ++j) {
    float pj = __shfl(p, j, 32);
    const bf16_t* vrow = vb + (blk * PBLK + 1 + j) * (size_t)DIM + h * DH;
    a0 += pj * (float)vrow[lane];
    a1 += pj * (float)vrow[lane + 32];
  }
  ctx[blk * DIM + h * DH + lane]      = (bf16_t)a0;
  ctx[blk * DIM + h * DH + 32 + lane] = (bf16_t)a1;
}

// ---------------------------------------------------------------------------
// memory attention: 1 wave per (b, h, 16-query tile). WMMA scores with K_mem
// as A operand (rows = memory slots) so each query column lives in a lane
// pair; flash-style online softmax accumulating sum(e) and sum(e * t_norm).
// Head mean via f32 global atomics into com_t[b][n]. Next K tile prefetched
// (global_prefetch_b8) while current tile's WMMA + softmax VALU runs.
// ---------------------------------------------------------------------------
__global__ __launch_bounds__(32) void attn_mem(const bf16_t* __restrict__ Q2,
                                               const bf16_t* __restrict__ Kmem,
                                               float* __restrict__ com_t) {
  const int bid = blockIdx.x;                     // b*64 + h*8 + nt
  const int nt = bid & 7, h = (bid >> 3) & 7, b = bid >> 6;
  const int lane = threadIdx.x;
  const int n0 = nt * 16;

  const bf16_t* q2b = Q2 + ((size_t)b * NBLK) * DIM + h * DH;
  v16bf bf0 = load_B_frag(q2b, DIM, n0, 0,  lane);
  v16bf bf1 = load_B_frag(q2b, DIM, n0, 32, lane);

  float m = -INFINITY, lsum = 0.0f, tacc = 0.0f;
  const int wlo = (lane < 16) ? 0 : 8;
  for (int wt = 0; wt < WL / 16; ++wt) {
    const int w0 = wt * 16;
    const bf16_t* kbase = Kmem + ((size_t)b * WL + w0) * DIM + h * DH;
    if (wt + 1 < WL / 16) {                       // prefetch next tile's rows
      __builtin_prefetch(kbase + (size_t)(16 + (lane & 15)) * DIM, 0, 1);
    }
    v8f c = {0,0,0,0,0,0,0,0};
    v16bf a0 = load_A_frag(kbase, DIM, 0, 0,  lane);
    v16bf a1 = load_A_frag(kbase, DIM, 0, 32, lane);
    c = __builtin_amdgcn_wmma_f32_16x16x32_bf16(false, a0, false, bf0,
                                                (short)0, c, false, false);
    c = __builtin_amdgcn_wmma_f32_16x16x32_bf16(false, a1, false, bf1,
                                                (short)0, c, false, false);
    float sc[8], tm = -INFINITY;
#pragma unroll
    for (int r = 0; r < 8; ++r) { sc[r] = c[r] * 0.125f; tm = fmaxf(tm, sc[r]); }
    tm = fmaxf(tm, __shfl_xor(tm, 16, 32));       // pair lanes share a query
    float mn = fmaxf(m, tm);
    float scale = __expf(m - mn);
    lsum *= scale; tacc *= scale;
#pragma unroll
    for (int r = 0; r < 8; ++r) {
      float e = __expf(sc[r] - mn);
      lsum += e;
      tacc += e * ((float)(w0 + wlo + r) * (1.0f / 1023.0f));  // t_norm
    }
    m = mn;
  }
  lsum += __shfl_xor(lsum, 16, 32);
  tacc += __shfl_xor(tacc, 16, 32);
  if (lane < 16) {
    int n = n0 + lane;
    atomicAdd(&com_t[(size_t)b * NBLK + n], (tacc / lsum) * (1.0f / HEADS));
  }
}

// ---------------------------------------------------------------------------
// output kernels (block-shifted scatter / broadcast)
// ---------------------------------------------------------------------------
__global__ void write_out1(const float* __restrict__ com_t, float* __restrict__ o) {
  int i = blockIdx.x * 256 + threadIdx.x;         // B*S
  int b = i >> 12, s = i & (SEQ - 1);
  int n = s >> 5;
  float v = ((s & 31) == 0 && n > 0) ? com_t[b * NBLK + n - 1] : 0.0f;
  o[i] = v;
}

__global__ void write_out2(const float* __restrict__ sumf, float* __restrict__ o) {
  size_t i = (size_t)blockIdx.x * 256 + threadIdx.x;  // B*S*D
  int c = (int)(i & 511);
  size_t row = i >> 9;
  int b = (int)(row >> 12), s = (int)(row & (SEQ - 1));
  int n = s >> 5;
  float v = (n > 0) ? sumf[((size_t)b * NBLK + n - 1) * DIM + c] : 0.0f;
  o[i] = v;
}

// ---------------------------------------------------------------------------
extern "C" void kernel_launch(void* const* d_in, const int* in_sizes, int n_in,
                              void* d_out, int out_size, void* d_ws, size_t ws_size,
                              hipStream_t stream) {
  (void)in_sizes; (void)n_in; (void)out_size; (void)ws_size;

  const float* y    = (const float*)d_in[0];
  const float* mem  = (const float*)d_in[1];
  // d_in[2..4] = spatial_shapes / level_start_index / bar_mask (compile-time)
  const float* Wq   = (const float*)d_in[5];
  const float* Wk   = (const float*)d_in[6];
  const float* Wv   = (const float*)d_in[7];
  const float* Wo   = (const float*)d_in[8];
  const float* Wq2  = (const float*)d_in[9];
  const float* Wkm  = (const float*)d_in[10];

  float* out1 = (float*)d_out;                          // B*S
  float* out2 = out1 + (size_t)BATCH * SEQ;             // B*S*D

  // ---- workspace carve-up (~128 MB) ----
  char* base = (char*)d_ws;
  size_t off = 0;
  auto carve = [&](size_t bytes) {
    off = (off + 255) & ~(size_t)255;
    char* p = base + off;
    off += bytes;
    return p;
  };
  const size_t ROWS = (size_t)BATCH * SEQ;              // 32768
  const size_t BN   = (size_t)BATCH * NBLK;             // 1024
  bf16_t* wt   = (bf16_t*)carve(6 * 262144 * sizeof(bf16_t));
  bf16_t* ype  = (bf16_t*)carve(ROWS * DIM * sizeof(bf16_t));
  bf16_t* aq   = (bf16_t*)carve(BN * DIM * sizeof(bf16_t));
  bf16_t* memb = (bf16_t*)carve((size_t)BATCH * WL * DIM * sizeof(bf16_t));
  bf16_t* qs   = (bf16_t*)carve(BN * DIM * sizeof(bf16_t));
  bf16_t* kb   = (bf16_t*)carve(ROWS * DIM * sizeof(bf16_t));
  bf16_t* vb   = (bf16_t*)carve(ROWS * DIM * sizeof(bf16_t));
  bf16_t* ctx  = (bf16_t*)carve(BN * DIM * sizeof(bf16_t));
  float*  sumf = (float*)carve(BN * DIM * sizeof(float));
  bf16_t* sumb = (bf16_t*)carve(BN * DIM * sizeof(bf16_t));
  bf16_t* q2   = (bf16_t*)carve(BN * DIM * sizeof(bf16_t));
  bf16_t* kmem = (bf16_t*)carve((size_t)BATCH * WL * DIM * sizeof(bf16_t));
  float*  comt = (float*)carve(BN * sizeof(float));

  // ---- prep ----
  prep_w  <<<6144,  256, 0, stream>>>(Wq, Wk, Wv, Wo, Wq2, Wkm, wt);
  prep_ype<<<65536, 256, 0, stream>>>(y, ype);
  prep_aq <<<2048,  256, 0, stream>>>(ype, aq);
  prep_mem<<<16384, 256, 0, stream>>>(mem, memb);
  zero_f32<<<(int)((BN + 255) / 256), 256, 0, stream>>>(comt, (int)BN);

  bf16_t* WqT  = wt + 0 * 262144;
  bf16_t* WkT  = wt + 1 * 262144;
  bf16_t* WvT  = wt + 2 * 262144;
  bf16_t* WoT  = wt + 3 * 262144;
  bf16_t* Wq2T = wt + 4 * 262144;
  bf16_t* WkmT = wt + 5 * 262144;

  // ---- projections (WMMA GEMMs), grid = (rows/32, 512/128) ----
  gemm_nt<<<dim3(32,   4), 128, 0, stream>>>(aq,  WqT, qs, nullptr, 0);  // q
  gemm_nt<<<dim3(1024, 4), 128, 0, stream>>>(ype, WkT, kb, nullptr, 0);  // K
  gemm_nt<<<dim3(1024, 4), 128, 0, stream>>>(ype, WvT, vb, nullptr, 0);  // V

  // ---- bar attention ----
  attn_bar<<<(int)BN, 256, 0, stream>>>(qs, kb, vb, ctx);

  // ---- summary / Q2 / K_mem ----
  gemm_nt<<<dim3(32,  4), 128, 0, stream>>>(ctx,  WoT,  sumb, sumf, 0);
  gemm_nt<<<dim3(32,  4), 128, 0, stream>>>(sumb, Wq2T, q2,   nullptr, 0);
  gemm_nt<<<dim3(256, 4), 128, 0, stream>>>(memb, WkmT, kmem, nullptr, 1);

  // ---- memory attention (flash-style, WMMA) ----
  attn_mem<<<BATCH * HEADS * (NBLK / 16), 32, 0, stream>>>(q2, kmem, comt);

  // ---- outputs ----
  write_out1<<<128,   256, 0, stream>>>(comt, out1);
  write_out2<<<65536, 256, 0, stream>>>(sumf, out2);
}
